// MeshGNN_90366111908154
// MI455X (gfx1250) — matrix operand order repaired and checked by reading, not compiled, not run on hardware
//
#include <hip/hip_runtime.h>
#include <hip/hip_bf16.h>
#include <math.h>

// ---------------------------------------------------------------------------
// MeshGNN forward for MI455X (gfx1250, wave32, WMMA bf16).
//
// Input index map (setup_inputs dict order; params flattened as jax pytree
// leaves = dict keys sorted alphabetically, lists in order):
//   0: x (16000x6 f32)   1: edge_index (2x96000 i32)   2: edge_attr (96000x4 f32)
//   3..8   dec:      W1, W2, W3, b1, b2, b3
//   9..14  edge_enc: W1, W2, b1, b2, be, g
//   15+12l layers[l]: edge{W1,W2,b1,b2,be,g}, node{W1,W2,b1,b2,be,g}
//   111..116 node_enc: W1, W2, b1, b2, be, g
//
// Blocks cover 32 rows (two 16-row M tiles); each of 8 waves owns a 16-col
// tile and keeps two f32 accumulators so every B fragment feeds 2 WMMAs.
// B fragments are loaded from a per-lane base pointer with compile-time
// byte offsets -> immediate-offset global_load_b128, no per-kb address math.
// ---------------------------------------------------------------------------

typedef __bf16 bf16_t;
typedef __attribute__((ext_vector_type(16))) __bf16        v16bf;
typedef __attribute__((ext_vector_type(8)))  float         v8f;
typedef __attribute__((ext_vector_type(4)))  unsigned int  u32x4;

union FragU { v16bf v; u32x4 u[2]; bf16_t e[16]; };

static __device__ __forceinline__ float siluf(float x) {
    return x * (1.0f / (1.0f + __expf(-x)));
}

// A fragment (16x32 bf16), ISA layout: lane row = lane&15; lanes<16 hold
// K = {k0..k0+7, k0+16..k0+23}, lanes>=16 hold K+8. Two b128 loads with
// immediate offsets from the row pointer.
static __device__ __forceinline__ v16bf afrag_bf(const bf16_t* rowp, int k0, int hi) {
    FragU f;
    f.u[0] = *(const u32x4*)(rowp + k0 + hi * 8);
    f.u[1] = *(const u32x4*)(rowp + k0 + 16 + hi * 8);
    return f.v;
}

// Same fragment but source rows are f32 (agg buffer): convert in-register.
static __device__ __forceinline__ v16bf afrag_f32(const float* rowp, int k0, int hi) {
    FragU f;
    const float* p0 = rowp + k0 + hi * 8;
    const float* p1 = rowp + k0 + 16 + hi * 8;
#pragma unroll
    for (int i = 0; i < 8; ++i) {
        f.e[i]     = (bf16_t)p0[i];
        f.e[8 + i] = (bf16_t)p1[i];
    }
    return f.v;
}

// B fragment from pre-swizzled weights: per-lane base pointer + element
// offset (compile-time in unrolled loops -> immediate vmem offsets).
static __device__ __forceinline__ v16bf bfrag(const bf16_t* lanep, int eoff) {
    FragU f;
    f.u[0] = *(const u32x4*)(lanep + eoff);
    f.u[1] = *(const u32x4*)(lanep + eoff + 8);
    return f.v;
}

static __device__ __forceinline__ v8f wmma_bf16(v16bf a, v16bf b, v8f c) {
    return __builtin_amdgcn_wmma_f32_16x16x32_bf16(false, a, false, b, (short)0, c, false, false);
}

// --------------------------- weight pre-swizzle ----------------------------
// W f32 (K x Ncols row-major) -> bf16 fragment order:
// dst[((kb*NT+nt)*32+lane)*16+i] = W[(kb*32+(i>>3)*16+(lane>>4)*8+(i&7)) * Ncols + nt*16+(lane&15)]
__global__ void k_swizzle_w(const float* __restrict__ W, bf16_t* __restrict__ dst,
                            int K, int Ncols) {
    int NT  = Ncols >> 4;
    int tid = blockIdx.x * blockDim.x + threadIdx.x;
    if (tid >= K * Ncols) return;
    int i    = tid & 15;
    int lane = (tid >> 4) & 31;
    int rest = tid >> 9;
    int nt   = rest % NT;
    int kb   = rest / NT;
    int k    = kb * 32 + (i >> 3) * 16 + (lane >> 4) * 8 + (i & 7);
    int col  = nt * 16 + (lane & 15);
    dst[tid] = (bf16_t)W[k * Ncols + col];
}

// --------------------- small-K first linear (encoders) ---------------------
__global__ void k_mlp1_small(const float* __restrict__ X, const float* __restrict__ W1,
                             const float* __restrict__ b1, bf16_t* __restrict__ out,
                             int M, int Kin) {
    int tid = blockIdx.x * blockDim.x + threadIdx.x;
    if (tid >= M * 128) return;
    int row = tid >> 7, col = tid & 127;
    float acc = b1[col];
    for (int k = 0; k < Kin; ++k) acc += X[row * Kin + k] * W1[k * 128 + col];
    out[tid] = (bf16_t)siluf(acc);
}

// -------------------- GEMM(128x128) + bias + LayerNorm ---------------------
// Block = 256 threads = 8 waves; 32 contiguous rows (two M tiles/wave).
// Optional residual add; writes f32 and/or bf16 outputs.
__global__ void __launch_bounds__(256) k_gemm_ln(
    const bf16_t* __restrict__ A, const bf16_t* __restrict__ Wswz,
    const float* __restrict__ bias, const float* __restrict__ g,
    const float* __restrict__ be, const float* __restrict__ resid,
    float* __restrict__ outF, bf16_t* __restrict__ outB) {
    __shared__ float tile[32][128];
    __shared__ float mu[32], rstd[32];
    int t = threadIdx.x, lane = t & 31, nt = t >> 5;
    int hi = lane >> 4, cl = lane & 15;
    long r0 = (long)blockIdx.x * 32;
    const bf16_t* a0 = A + (r0 + cl) * 128;
    const bf16_t* a1 = A + (r0 + 16 + cl) * 128;
    const bf16_t* wl = Wswz + (nt * 32 + lane) * 16;
    v8f c0 = {}, c1 = {};
#pragma unroll
    for (int kb = 0; kb < 4; ++kb) {
        v16bf b = bfrag(wl, kb * 4096);  // NT=8: kb stride = 8*32*16 elems
        c0 = wmma_bf16(afrag_bf(a0, kb * 32, hi), b, c0);
        c1 = wmma_bf16(afrag_bf(a1, kb * 32, hi), b, c1);
    }
    int col = nt * 16 + cl;
    float bb = bias[col];
#pragma unroll
    for (int i = 0; i < 8; ++i) {
        tile[hi * 8 + i][col]      = c0[i] + bb;
        tile[16 + hi * 8 + i][col] = c1[i] + bb;
    }
    __syncthreads();
    if (t < 32) {
        float s = 0.f, s2 = 0.f;
        for (int j = 0; j < 128; ++j) { float v = tile[t][j]; s += v; s2 += v * v; }
        float m = s * (1.0f / 128.0f);
        mu[t]   = m;
        rstd[t] = rsqrtf(s2 * (1.0f / 128.0f) - m * m + 1e-5f);
    }
    __syncthreads();
#pragma unroll
    for (int j = 0; j < 16; ++j) {
        int idx = t * 16 + j, rr = idx >> 7, cc = idx & 127;
        float v = (tile[rr][cc] - mu[rr]) * rstd[rr] * g[cc] + be[cc];
        long gi = (r0 + rr) * 128 + cc;
        if (resid) v += resid[gi];
        if (outF) outF[gi] = v;
        if (outB) outB[gi] = (bf16_t)v;
    }
}

// ------------------- edge MLP layer 1: gather-GEMM + SiLU ------------------
// msg1 = silu(h[dst]@W1a + h[src]@W1b + e@W1c + b1); W1 swizzled as 12 kb.
__global__ void __launch_bounds__(256) k_edge_mlp1(
    const bf16_t* __restrict__ hB, const bf16_t* __restrict__ eB,
    const int* __restrict__ srcIdx, const int* __restrict__ dstIdx,
    const bf16_t* __restrict__ W1swz, const float* __restrict__ b1,
    bf16_t* __restrict__ msg1) {
    __shared__ int rs[32], rd[32];
    int t = threadIdx.x;
    long e0 = (long)blockIdx.x * 32;
    if (t < 32) rd[t] = dstIdx[e0 + t];
    else if (t < 64) rs[t - 32] = srcIdx[e0 + (t - 32)];
    __syncthreads();
    int lane = t & 31, nt = t >> 5, hi = lane >> 4, cl = lane & 15;
    const bf16_t* wl  = W1swz + (nt * 32 + lane) * 16;
    const bf16_t* aD0 = hB + (long)rd[cl] * 128;
    const bf16_t* aD1 = hB + (long)rd[16 + cl] * 128;
    const bf16_t* aS0 = hB + (long)rs[cl] * 128;
    const bf16_t* aS1 = hB + (long)rs[16 + cl] * 128;
    const bf16_t* aE0 = eB + (e0 + cl) * 128;
    const bf16_t* aE1 = eB + (e0 + 16 + cl) * 128;
    v8f c0 = {}, c1 = {};
#pragma unroll
    for (int kb = 0; kb < 4; ++kb) {
        v16bf b = bfrag(wl, kb * 4096);
        c0 = wmma_bf16(afrag_bf(aD0, kb * 32, hi), b, c0);
        c1 = wmma_bf16(afrag_bf(aD1, kb * 32, hi), b, c1);
    }
#pragma unroll
    for (int kb = 0; kb < 4; ++kb) {
        v16bf b = bfrag(wl, (4 + kb) * 4096);
        c0 = wmma_bf16(afrag_bf(aS0, kb * 32, hi), b, c0);
        c1 = wmma_bf16(afrag_bf(aS1, kb * 32, hi), b, c1);
    }
#pragma unroll
    for (int kb = 0; kb < 4; ++kb) {
        v16bf b = bfrag(wl, (8 + kb) * 4096);
        c0 = wmma_bf16(afrag_bf(aE0, kb * 32, hi), b, c0);
        c1 = wmma_bf16(afrag_bf(aE1, kb * 32, hi), b, c1);
    }
    int col = nt * 16 + cl;
    float bb = b1[col];
#pragma unroll
    for (int i = 0; i < 8; ++i) {
        msg1[(e0 + hi * 8 + i) * 128 + col]      = (bf16_t)siluf(c0[i] + bb);
        msg1[(e0 + 16 + hi * 8 + i) * 128 + col] = (bf16_t)siluf(c1[i] + bb);
    }
}

// ------- edge MLP layer 2: GEMM + LN, fused mean-scatter into agg ----------
__global__ void __launch_bounds__(256) k_edge_mlp2_scatter(
    const bf16_t* __restrict__ msg1, const bf16_t* __restrict__ W2swz,
    const float* __restrict__ b2, const float* __restrict__ g,
    const float* __restrict__ be, const int* __restrict__ dstIdx,
    const float* __restrict__ inv, float* __restrict__ agg) {
    __shared__ float tile[32][128];
    __shared__ float mu[32], rstd[32];
    __shared__ int rd[32];
    int t = threadIdx.x;
    long e0 = (long)blockIdx.x * 32;
    if (t < 32) rd[t] = dstIdx[e0 + t];
    int lane = t & 31, nt = t >> 5, hi = lane >> 4, cl = lane & 15;
    const bf16_t* a0 = msg1 + (e0 + cl) * 128;
    const bf16_t* a1 = msg1 + (e0 + 16 + cl) * 128;
    const bf16_t* wl = W2swz + (nt * 32 + lane) * 16;
    v8f c0 = {}, c1 = {};
#pragma unroll
    for (int kb = 0; kb < 4; ++kb) {
        v16bf b = bfrag(wl, kb * 4096);
        c0 = wmma_bf16(afrag_bf(a0, kb * 32, hi), b, c0);
        c1 = wmma_bf16(afrag_bf(a1, kb * 32, hi), b, c1);
    }
    int col = nt * 16 + cl;
    float bb = b2[col];
#pragma unroll
    for (int i = 0; i < 8; ++i) {
        tile[hi * 8 + i][col]      = c0[i] + bb;
        tile[16 + hi * 8 + i][col] = c1[i] + bb;
    }
    __syncthreads();
    if (t < 32) {
        float s = 0.f, s2 = 0.f;
        for (int j = 0; j < 128; ++j) { float v = tile[t][j]; s += v; s2 += v * v; }
        float m = s * (1.0f / 128.0f);
        mu[t]   = m;
        rstd[t] = rsqrtf(s2 * (1.0f / 128.0f) - m * m + 1e-5f);
    }
    __syncthreads();
#pragma unroll
    for (int j = 0; j < 16; ++j) {
        int idx = t * 16 + j, rr = idx >> 7, cc = idx & 127;
        int d = rd[rr];
        float v = (tile[rr][cc] - mu[rr]) * rstd[rr] * g[cc] + be[cc];
        atomicAdd(&agg[(long)d * 128 + cc], v * inv[d]);
    }
}

// --------------- node MLP layer 1: [h | agg] @ W1 + b1, SiLU ---------------
__global__ void __launch_bounds__(256) k_node_mlp1(
    const bf16_t* __restrict__ hB, const float* __restrict__ agg,
    const bf16_t* __restrict__ W1swz, const float* __restrict__ b1,
    bf16_t* __restrict__ outB) {
    int t = threadIdx.x, lane = t & 31, nt = t >> 5, hi = lane >> 4, cl = lane & 15;
    long r0 = (long)blockIdx.x * 32;
    const bf16_t* ah0 = hB + (r0 + cl) * 128;
    const bf16_t* ah1 = hB + (r0 + 16 + cl) * 128;
    const float*  aa0 = agg + (r0 + cl) * 128;
    const float*  aa1 = agg + (r0 + 16 + cl) * 128;
    const bf16_t* wl  = W1swz + (nt * 32 + lane) * 16;
    v8f c0 = {}, c1 = {};
#pragma unroll
    for (int kb = 0; kb < 4; ++kb) {
        v16bf b = bfrag(wl, kb * 4096);
        c0 = wmma_bf16(afrag_bf(ah0, kb * 32, hi), b, c0);
        c1 = wmma_bf16(afrag_bf(ah1, kb * 32, hi), b, c1);
    }
#pragma unroll
    for (int kb = 0; kb < 4; ++kb) {
        v16bf b = bfrag(wl, (4 + kb) * 4096);
        c0 = wmma_bf16(afrag_f32(aa0, kb * 32, hi), b, c0);
        c1 = wmma_bf16(afrag_f32(aa1, kb * 32, hi), b, c1);
    }
    int col = nt * 16 + cl;
    float bb = b1[col];
#pragma unroll
    for (int i = 0; i < 8; ++i) {
        outB[(r0 + hi * 8 + i) * 128 + col]      = (bf16_t)siluf(c0[i] + bb);
        outB[(r0 + 16 + hi * 8 + i) * 128 + col] = (bf16_t)siluf(c1[i] + bb);
    }
}

// --------------------- generic GEMM + SiLU (decoder) -----------------------
// blockDim = 32*NT, 32 rows per block, K = 128 (4 kb).
__global__ void k_gemm_silu(const bf16_t* __restrict__ A, const bf16_t* __restrict__ Wswz,
                            const float* __restrict__ bias, bf16_t* __restrict__ out,
                            int NT) {
    int t = threadIdx.x, lane = t & 31, nt = t >> 5, hi = lane >> 4, cl = lane & 15;
    long r0 = (long)blockIdx.x * 32;
    const bf16_t* a0 = A + (r0 + cl) * 128;
    const bf16_t* a1 = A + (r0 + 16 + cl) * 128;
    const bf16_t* wl = Wswz + (nt * 32 + lane) * 16;
    int kstride = NT * 512;  // NT*32*16 elems per kb
    v8f c0 = {}, c1 = {};
#pragma unroll
    for (int kb = 0; kb < 4; ++kb) {
        v16bf b = bfrag(wl, kb * kstride);
        c0 = wmma_bf16(afrag_bf(a0, kb * 32, hi), b, c0);
        c1 = wmma_bf16(afrag_bf(a1, kb * 32, hi), b, c1);
    }
    int col = nt * 16 + cl, ldo = NT * 16;
    float bb = bias[col];
#pragma unroll
    for (int i = 0; i < 8; ++i) {
        out[(r0 + hi * 8 + i) * ldo + col]      = (bf16_t)siluf(c0[i] + bb);
        out[(r0 + 16 + hi * 8 + i) * ldo + col] = (bf16_t)siluf(c1[i] + bb);
    }
}

// ------------------------------ small kernels ------------------------------
__global__ void k_zero(float* __restrict__ p, long n) {
    long i = (long)blockIdx.x * blockDim.x + threadIdx.x;
    if (i < n) p[i] = 0.f;
}
__global__ void k_count(const int* __restrict__ dstIdx, float* __restrict__ cnt, int E) {
    int i = blockIdx.x * blockDim.x + threadIdx.x;
    if (i < E) atomicAdd(&cnt[dstIdx[i]], 1.0f);
}
__global__ void k_inv(const float* __restrict__ cnt, float* __restrict__ inv, int N) {
    int i = blockIdx.x * blockDim.x + threadIdx.x;
    if (i < N) inv[i] = 1.0f / fmaxf(cnt[i], 1.0f);
}
__global__ void k_dec3(const bf16_t* __restrict__ z2, const float* __restrict__ W3,
                       const float* __restrict__ b3, float* __restrict__ out, int N) {
    int tid = blockIdx.x * blockDim.x + threadIdx.x;
    if (tid >= N * 4) return;
    int row = tid >> 2, o = tid & 3;
    float acc = b3[o];
    for (int k = 0; k < 64; ++k) acc += (float)z2[row * 64 + k] * W3[k * 4 + o];
    out[tid] = acc;
}

// ---------------------------------------------------------------------------
extern "C" void kernel_launch(void* const* d_in, const int* in_sizes, int n_in,
                              void* d_out, int out_size, void* d_ws, size_t ws_size,
                              hipStream_t stream) {
    const int N = 16000, E = 96000;
    const float* x  = (const float*)d_in[0];
    const int*   ei = (const int*)d_in[1];
    const float* ea = (const float*)d_in[2];
    const int* srcIdx = ei;
    const int* dstIdx = ei + E;
    auto F = [&](int i) { return (const float*)d_in[i]; };
    // dec
    const float *dW1 = F(3), *dW2 = F(4), *dW3 = F(5), *db1 = F(6), *db2 = F(7), *db3 = F(8);
    // edge_enc / node_enc
    const float *eeW1 = F(9), *eeW2 = F(10), *eeb1 = F(11), *eeb2 = F(12), *eebe = F(13), *eeg = F(14);
    const float *neW1 = F(111), *neW2 = F(112), *neb1 = F(113), *neb2 = F(114), *nebe = F(115), *neg = F(116);

    // ---- workspace carve (deterministic) ----
    char* w = (char*)d_ws;
    auto carve = [&](size_t bytes) { void* p = (void*)w; w += (bytes + 255) & ~(size_t)255; return p; };
    bf16_t* sw_neW2 = (bf16_t*)carve(128 * 128 * 2);
    bf16_t* sw_eeW2 = (bf16_t*)carve(128 * 128 * 2);
    bf16_t* sw_dW1  = (bf16_t*)carve(128 * 128 * 2);
    bf16_t* sw_dW2  = (bf16_t*)carve(128 * 64 * 2);
    bf16_t *sw_eW1[8], *sw_eW2[8], *sw_nW1[8], *sw_nW2[8];
    for (int l = 0; l < 8; ++l) {
        sw_eW1[l] = (bf16_t*)carve(384 * 128 * 2);
        sw_eW2[l] = (bf16_t*)carve(128 * 128 * 2);
        sw_nW1[l] = (bf16_t*)carve(256 * 128 * 2);
        sw_nW2[l] = (bf16_t*)carve(128 * 128 * 2);
    }
    float*  hF   = (float*)carve((size_t)N * 128 * 4);
    bf16_t* hB   = (bf16_t*)carve((size_t)N * 128 * 2);
    bf16_t* eB   = (bf16_t*)carve((size_t)E * 128 * 2);
    bf16_t* msg1 = (bf16_t*)carve((size_t)E * 128 * 2);
    float*  agg  = (float*)carve((size_t)N * 128 * 4);
    bf16_t* tbf  = (bf16_t*)carve((size_t)N * 128 * 2);
    float*  cnt  = (float*)carve((size_t)N * 4);
    float*  inv  = (float*)carve((size_t)N * 4);
    bf16_t* z1   = tbf;             // reuse after layers
    bf16_t* z2   = (bf16_t*)msg1;   // reuse after layers

    auto swz = [&](const float* W, bf16_t* dst, int K, int Ncols) {
        int total = K * Ncols;
        k_swizzle_w<<<(total + 255) / 256, 256, 0, stream>>>(W, dst, K, Ncols);
    };

    // 1) weight prep (bf16 + fragment swizzle)
    swz(neW2, sw_neW2, 128, 128);
    swz(eeW2, sw_eeW2, 128, 128);
    swz(dW1, sw_dW1, 128, 128);
    swz(dW2, sw_dW2, 128, 64);
    for (int l = 0; l < 8; ++l) {
        int b = 15 + 12 * l;
        swz(F(b + 0), sw_eW1[l], 384, 128);
        swz(F(b + 1), sw_eW2[l], 128, 128);
        swz(F(b + 6), sw_nW1[l], 256, 128);
        swz(F(b + 7), sw_nW2[l], 128, 128);
    }

    // 2) degree -> inverse count
    k_zero<<<(N + 255) / 256, 256, 0, stream>>>(cnt, N);
    k_count<<<(E + 255) / 256, 256, 0, stream>>>(dstIdx, cnt, E);
    k_inv<<<(N + 255) / 256, 256, 0, stream>>>(cnt, inv, N);

    // 3) encoders
    k_mlp1_small<<<(N * 128 + 255) / 256, 256, 0, stream>>>(x, neW1, neb1, tbf, N, 6);
    k_gemm_ln<<<N / 32, 256, 0, stream>>>(tbf, sw_neW2, neb2, neg, nebe,
                                          nullptr, hF, hB);
    k_mlp1_small<<<(E * 128 + 255) / 256, 256, 0, stream>>>(ea, eeW1, eeb1, msg1, E, 4);
    k_gemm_ln<<<E / 32, 256, 0, stream>>>(msg1, sw_eeW2, eeb2, eeg, eebe,
                                          nullptr, nullptr, eB);

    // 4) message-passing layers
    for (int l = 0; l < 8; ++l) {
        int b = 15 + 12 * l;
        const float *eb1 = F(b + 2), *eb2 = F(b + 3), *ebe = F(b + 4), *eg = F(b + 5);
        const float *nb1 = F(b + 8), *nb2 = F(b + 9), *nbe = F(b + 10), *ng = F(b + 11);
        k_zero<<<((long)N * 128 + 255) / 256, 256, 0, stream>>>(agg, (long)N * 128);
        k_edge_mlp1<<<E / 32, 256, 0, stream>>>(hB, eB, srcIdx, dstIdx, sw_eW1[l], eb1, msg1);
        k_edge_mlp2_scatter<<<E / 32, 256, 0, stream>>>(msg1, sw_eW2[l], eb2, eg, ebe,
                                                        dstIdx, inv, agg);
        k_node_mlp1<<<N / 32, 256, 0, stream>>>(hB, agg, sw_nW1[l], nb1, tbf);
        k_gemm_ln<<<N / 32, 256, 0, stream>>>(tbf, sw_nW2[l], nb2, ng, nbe,
                                              hF /*resid*/, hF, hB);
    }

    // 5) decoder
    k_gemm_silu<<<N / 32, 256, 0, stream>>>(hB, sw_dW1, db1, z1, 8);
    k_gemm_silu<<<N / 32, 128, 0, stream>>>(z1, sw_dW2, db2, z2, 4);
    k_dec3<<<(N * 4 + 255) / 256, 256, 0, stream>>>(z2, dW3, db3, (float*)d_out, N);
}
